// LoraLSTM_54614804136074
// MI455X (gfx1250) — matrix operand order repaired and checked by reading, not compiled, not run on hardware
//
#include <hip/hip_runtime.h>
#include <math.h>

#define DEVINL __device__ __forceinline__

typedef __attribute__((ext_vector_type(16))) __bf16 v16bf;
typedef __attribute__((ext_vector_type(8)))  float  v8f;

struct alignas(16) U4 { unsigned int x, y, z, w; };
struct U8 { U4 a, b; };

static constexpr int kB  = 32;    // batch
static constexpr int kT  = 512;   // seq len
static constexpr int kI  = 1024;  // input dim (both layers: INP == 2H)
static constexpr int kH  = 512;   // hidden
static constexpr int kG  = 2048;  // 4*H gates
static constexpr int kR  = 8;     // LoRA rank
static constexpr int kBT = kB * kT;
static constexpr float kScale = 2.0f;  // ALPHA / R

DEVINL unsigned short f2bf(float f) {
  unsigned int u = __builtin_bit_cast(unsigned int, f);
  return (unsigned short)((u + 0x7FFFu + ((u >> 16) & 1u)) >> 16);  // RNE
}
DEVINL float bf2f(unsigned short h) {
  unsigned int u = ((unsigned int)h) << 16;
  return __builtin_bit_cast(float, u);
}
DEVINL float sigmoidf_(float x) { return 1.0f / (1.0f + __expf(-x)); }

// A fragment (16x32 bf16, MxK): lane m = lane&15, half = lane>>4.
// Per ISA layout: VGPR0-3 hold K = 8*half + 0..7, VGPR4-7 hold K = 16 + 8*half + 0..7.
DEVINL v16bf loadA16x32(const unsigned short* row, int lh) {
  U8 p;
  p.a = *(const U4*)(row + 8 * lh);
  p.b = *(const U4*)(row + 16 + 8 * lh);
  return __builtin_bit_cast(v16bf, p);
}
// B fragment (32x16 bf16, KxN): lane n = lane&15, K = 8*half+0..7, 16+8*half+0..7.
// 'p' must point at wT[k0][n0 + (lane&15)] of a K-major (transposed) weight.
DEVINL v16bf loadB32x16(const unsigned short* p, int lh) {
  v16bf q;
#pragma unroll
  for (int s = 0; s < 8; ++s) {
    q[s] = __builtin_bit_cast(__bf16, p[(size_t)(8 * lh + s) * kG]);
    q[8 + s] = __builtin_bit_cast(__bf16, p[(size_t)(16 + 8 * lh + s) * kG]);
  }
  return q;
}
DEVINL v8f wmma_bf16(v16bf a, v16bf b, v8f c) {
  return __builtin_amdgcn_wmma_f32_16x16x32_bf16(false, a, false, b, (short)0, c,
                                                 false, false);
}
DEVINL float dot8(const float* __restrict__ a, const float* __restrict__ b) {
  float s = 0.f;
#pragma unroll
  for (int i = 0; i < 8; ++i) s += a[i] * b[i];
  return s;
}

// ---------------------------------------------------------------- prep kernels
__global__ void k_cvt_bf16(const float* __restrict__ src,
                           unsigned short* __restrict__ dst, size_t n) {
  size_t i = (size_t)blockIdx.x * blockDim.x + threadIdx.x;
  size_t st = (size_t)gridDim.x * blockDim.x;
  for (; i < n; i += st) dst[i] = f2bf(src[i]);
}

// w: (nc, nrows, ncols) f32  ->  wT: (nc, ncols, nrows) bf16   (K-major weights)
__global__ void k_transpose_bf16(const float* __restrict__ w,
                                 unsigned short* __restrict__ wT,
                                 int nc, int nrows, int ncols) {
  size_t n = (size_t)nc * nrows * ncols;
  size_t i = (size_t)blockIdx.x * blockDim.x + threadIdx.x;
  size_t st = (size_t)gridDim.x * blockDim.x;
  for (; i < n; i += st) {
    int k = (int)(i % ncols);
    size_t t = i / ncols;
    int r = (int)(t % nrows);
    int c = (int)(t / nrows);
    wT[((size_t)c * ncols + k) * nrows + r] = f2bf(w[i]);
  }
}

__global__ void k_bias(const float* __restrict__ a, const float* __restrict__ b,
                       float* __restrict__ o, int n) {
  int i = blockIdx.x * blockDim.x + threadIdx.x;
  if (i < n) o[i] = a[i] + b[i];
}

// u[cl][m][r] = x_row(m) . A[cl][r][:],  m = t*32 + b  (matches gx GEMM M order)
__global__ void k_lora_u(const float* __restrict__ srcF,
                         const unsigned short* __restrict__ srcB, int isBF,
                         const float* __restrict__ A, float* __restrict__ u, int K) {
  int id = blockIdx.x * blockDim.x + threadIdx.x;
  if (id >= 2 * kBT * kR) return;
  int r = id & 7;
  int m = (id >> 3) & (kBT - 1);
  int cl = id >> 17;
  int b = m & 31, t = m >> 5;
  const float* ar = A + ((size_t)cl * kR + r) * K;
  float acc = 0.f;
  if (isBF) {
    const unsigned short* row = srcB + ((size_t)b * kT + t) * K;
    for (int k = 0; k < K; ++k) acc += bf2f(row[k]) * ar[k];
  } else {
    const float* row = srcF + ((size_t)b * kT + t) * K;
    for (int k = 0; k < K; ++k) acc += row[k] * ar[k];
  }
  u[((size_t)cl * kBT + m) * kR + r] = acc;
}

// ------------------------------------------------- time-parallel gx WMMA GEMM
// gx[cl][t][b][n] = sum_k x[b][t][k]*W[cl][n][k] + bias[cl][n]
//                 + 2 * dot8(u[cl][m], bl[cl][n])        (m = t*32 + b)
__global__ void __launch_bounds__(256) k_gx_gemm(
    const unsigned short* __restrict__ Abf,  // (B,T,K) bf16
    const unsigned short* __restrict__ WT,   // (2,K,G) bf16 (pair, K-major)
    const float* __restrict__ bias,          // (2,G)
    const float* __restrict__ u,             // (2,BT,R)
    const float* __restrict__ bl,            // (2,G,R)
    float* __restrict__ gx,                  // (2,T,B,G)
    int K) {
  int wid = (int)(((size_t)blockIdx.x * blockDim.x + threadIdx.x) >> 5);
  int lane = threadIdx.x & 31;
  int ng = wid & 31;           // 32 n-groups of 64 columns
  int mt = (wid >> 5) & 1023;  // 1024 m-tiles of 16 rows
  int cl = wid >> 15;          // cell within pair
  int ln = lane & 15, lh = lane >> 4;
  int n0 = ng * 64;

  const unsigned short* wt = WT + (size_t)cl * K * kG;
  int mrow = mt * 16 + ln;     // this lane's A row
  int b = mrow & 31, t = mrow >> 5;
  const unsigned short* arow = Abf + ((size_t)b * kT + t) * K;

  v8f acc0 = {}, acc1 = {}, acc2 = {}, acc3 = {};
  for (int k0 = 0; k0 < K; k0 += 32) {
    v16bf a = loadA16x32(arow + k0, lh);
    const unsigned short* wrow = wt + (size_t)k0 * kG + n0 + ln;
    v16bf b0 = loadB32x16(wrow, lh);
    v16bf b1 = loadB32x16(wrow + 16, lh);
    v16bf b2 = loadB32x16(wrow + 32, lh);
    v16bf b3 = loadB32x16(wrow + 48, lh);
    acc0 = wmma_bf16(a, b0, acc0);
    acc1 = wmma_bf16(a, b1, acc1);
    acc2 = wmma_bf16(a, b2, acc2);
    acc3 = wmma_bf16(a, b3, acc3);
  }
  const float* biasC = bias + (size_t)cl * kG;
  const float* blC = bl + (size_t)cl * kG * kR;
  const float* uC = u + (size_t)cl * kBT * kR;
  float* gxC = gx + (size_t)cl * kT * kB * kG;
  v8f accs[4] = {acc0, acc1, acc2, acc3};
#pragma unroll
  for (int j = 0; j < 4; ++j) {
    int n = n0 + 16 * j + ln;
    float bv = biasC[n];
    const float* blr = blC + (size_t)n * kR;
#pragma unroll
    for (int v = 0; v < 8; ++v) {
      int m = mt * 16 + 8 * lh + v;   // C/D layout: m = v + 8*(lane/16)
      int bb = m & 31, tt = m >> 5;
      float lora = dot8(uC + (size_t)m * kR, blr);
      gxC[((size_t)tt * kB + bb) * kG + n] = accs[j][v] + bv + kScale * lora;
    }
  }
}

// ---------------------------------------------- persistent LDS-resident scan
// One workgroup per direction. 16 waves; wave w owns hidden units [32w,32w+32)
// for both batch halves and computes its own i/f/g/o tiles -> no gate exchange.
__global__ void __launch_bounds__(512) k_lstm_scan(
    const float* __restrict__ gx,           // (2,T,B,G)
    const unsigned short* __restrict__ WT,  // (2,H,G) bf16 w_hh^T pair
    const float* __restrict__ Ahh,          // (2,R,H)
    const float* __restrict__ Bl,           // (2,G,R)
    unsigned short* __restrict__ ybf,       // (B,T,2H) bf16 layer output
    float* __restrict__ y32,                // optional (B,T,2H) f32 (layer 1)
    float* __restrict__ hn,                 // (2,B,H)
    float* __restrict__ cn) {               // (2,B,H)
  const int d = blockIdx.x;
  extern __shared__ char smraw[];
  float* hf = (float*)smraw;                              // 32*512 f32
  float* cf = hf + kB * kH;                               // 32*512 f32
  unsigned short* hb0 = (unsigned short*)(cf + kB * kH);  // 32*512 bf16
  unsigned short* hb1 = hb0 + kB * kH;                    // double buffer
  float* up = (float*)(hb1 + kB * kH);                    // 32*8*2 partials
  float* uu = up + kB * kR * 2;                           // 32*8

  const int tid = threadIdx.x;
  for (int i = tid; i < kB * kH; i += 512) {
    hf[i] = 0.f; cf[i] = 0.f; hb0[i] = 0; hb1[i] = 0;
  }
  __syncthreads();

  const unsigned short* wt = WT + (size_t)d * kH * kG;
  const float* ah = Ahh + (size_t)d * kR * kH;
  const float* bl = Bl + (size_t)d * kG * kR;
  const float* gxd = gx + (size_t)d * kT * kB * kG;

  const int lane = tid & 31;
  const int w = tid >> 5;  // 16 waves
  const int ln = lane & 15, lh = lane >> 4;
  const bool rev = (d == 1);

  for (int s = 0; s < kT; ++s) {
    const int t = rev ? (kT - 1 - s) : s;
    const unsigned short* hcur = (s & 1) ? hb1 : hb0;
    unsigned short* hnxt = (s & 1) ? hb0 : hb1;

    // ---- recurrent LoRA: uu[b][r] = h . a_hh[r] (rank 8, from f32 h) ----
    {
      int b = tid >> 4, r = (tid >> 1) & 7, q = tid & 1;
      const float* hr = hf + b * kH + q * 256;
      const float* ar = ah + r * kH + q * 256;
      float acc = 0.f;
      for (int k = 0; k < 256; ++k) acc += hr[k] * ar[k];
      up[tid] = acc;  // up[b*16 + r*2 + q]
    }
    __syncthreads();
    if (tid < kB * kR) uu[tid] = up[tid * 2] + up[tid * 2 + 1];
    __syncthreads();

    const float* gxt = gxd + (size_t)t * kB * kG;

#pragma unroll 1
    for (int gg = 0; gg < 4; ++gg) {       // 4 tile-groups per wave
      int G2 = w * 4 + gg;
      int mt = G2 & 1;                     // batch half
      int j0 = (G2 >> 1) * 16;             // hidden-unit tile
      v8f ai = {}, af = {}, ag = {}, ao = {};
      for (int kk = 0; kk < kH; kk += 32) {
        const unsigned short* arow = hcur + (mt * 16 + ln) * kH + kk;
        v16bf A = loadA16x32(arow, lh);
        const unsigned short* wrow = wt + (size_t)kk * kG + j0 + ln;
        __builtin_prefetch(wrow + (size_t)32 * kG, 0, 1);
        v16bf Bi = loadB32x16(wrow, lh);
        v16bf Bf = loadB32x16(wrow + kH, lh);
        v16bf Bg = loadB32x16(wrow + 2 * kH, lh);
        v16bf Bo = loadB32x16(wrow + 3 * kH, lh);
        ai = wmma_bf16(A, Bi, ai);
        af = wmma_bf16(A, Bf, af);
        ag = wmma_bf16(A, Bg, ag);
        ao = wmma_bf16(A, Bo, ao);
      }
      // ---- pointwise cell update for this tile (exclusive ownership) ----
      int n = j0 + ln;
      const float* bli = bl + (size_t)(0 * kH + n) * kR;
      const float* blf = bl + (size_t)(1 * kH + n) * kR;
      const float* blg = bl + (size_t)(2 * kH + n) * kR;
      const float* blo = bl + (size_t)(3 * kH + n) * kR;
#pragma unroll
      for (int v = 0; v < 8; ++v) {
        int b = mt * 16 + 8 * lh + v;
        const float* ur = uu + b * kR;
        float vi = ai[v] + gxt[(size_t)b * kG + 0 * kH + n] + kScale * dot8(ur, bli);
        float vf = af[v] + gxt[(size_t)b * kG + 1 * kH + n] + kScale * dot8(ur, blf);
        float vg = ag[v] + gxt[(size_t)b * kG + 2 * kH + n] + kScale * dot8(ur, blg);
        float vo = ao[v] + gxt[(size_t)b * kG + 3 * kH + n] + kScale * dot8(ur, blo);
        float iv = sigmoidf_(vi), fv = sigmoidf_(vf);
        float gv = tanhf(vg), ov = sigmoidf_(vo);
        float cold = cf[b * kH + n];
        float cnew = fv * cold + iv * gv;
        float hnew = ov * tanhf(cnew);
        cf[b * kH + n] = cnew;
        hf[b * kH + n] = hnew;
        hnxt[b * kH + n] = f2bf(hnew);
        size_t yo = ((size_t)b * kT + t) * (2 * kH) + (size_t)d * kH + n;
        ybf[yo] = f2bf(hnew);
        if (y32) y32[yo] = hnew;
      }
    }
    __syncthreads();  // h_{t+1} (hf/hnxt) complete before next step's reads
  }

  for (int i = tid; i < kB * kH; i += 512) {
    int b = i >> 9, j = i & (kH - 1);
    hn[((size_t)d * kB + b) * kH + j] = hf[i];
    cn[((size_t)d * kB + b) * kH + j] = cf[i];
  }
}

// ------------------------------------------------------------------ dispatch
extern "C" void kernel_launch(void* const* d_in, const int* in_sizes, int n_in,
                              void* d_out, int out_size, void* d_ws, size_t ws_size,
                              hipStream_t stream) {
  (void)in_sizes; (void)n_in; (void)out_size; (void)ws_size;
  const float* x     = (const float*)d_in[0];
  const float* w_ih  = (const float*)d_in[1];
  const float* w_hh  = (const float*)d_in[2];
  const float* b_ih  = (const float*)d_in[3];
  const float* b_hh  = (const float*)d_in[4];
  const float* a_ih  = (const float*)d_in[5];
  const float* bl_ih = (const float*)d_in[6];
  const float* a_hh  = (const float*)d_in[7];
  const float* bl_hh = (const float*)d_in[8];

  char* ws = (char*)d_ws;
  size_t off = 0;
  auto carve = [&](size_t bytes) -> void* {
    void* p = ws + off;
    off = (off + bytes + 255) & ~(size_t)255;
    return p;
  };
  unsigned short* xb   = (unsigned short*)carve((size_t)kBT * kI * 2);      // 32 MB
  unsigned short* wihT = (unsigned short*)carve((size_t)4 * kI * kG * 2);   // 16 MB
  unsigned short* whhT = (unsigned short*)carve((size_t)4 * kH * kG * 2);   //  8 MB
  float* bias  = (float*)carve((size_t)4 * kG * 4);
  float* u     = (float*)carve((size_t)4 * kBT * kR * 4);                   //  2 MB
  unsigned short* ybf = (unsigned short*)carve((size_t)kBT * kI * 2);       // 32 MB
  float* gxbuf = (float*)carve((size_t)2 * kT * kB * kG * 4);               // 256 MB

  float* out32 = (float*)d_out;                         // (B,T,2H)
  float* hn = out32 + (size_t)kB * kT * (2 * kH);       // (4,B,H)
  float* cn = hn + (size_t)4 * kB * kH;                 // (4,B,H)

  // prep
  k_cvt_bf16<<<4096, 256, 0, stream>>>(x, xb, (size_t)kBT * kI);
  k_transpose_bf16<<<4096, 256, 0, stream>>>(w_ih, wihT, 4, kG, kI);
  k_transpose_bf16<<<4096, 256, 0, stream>>>(w_hh, whhT, 4, kG, kH);
  k_bias<<<(4 * kG + 255) / 256, 256, 0, stream>>>(b_ih, b_hh, bias, 4 * kG);

  size_t smem = (size_t)kB * kH * 4 * 2   // hf, cf
              + (size_t)kB * kH * 2 * 2   // hb0, hb1
              + (size_t)kB * kR * 2 * 4   // up
              + (size_t)kB * kR * 4;      // uu   => ~200 KB (< 320 KB/WGP)

  // ---- layer 0 (cells 0,1): input = x ----
  k_lora_u<<<(2 * kBT * kR) / 256, 256, 0, stream>>>(x, (const unsigned short*)nullptr,
                                                     0, a_ih, u, kI);
  k_gx_gemm<<<8192, 256, 0, stream>>>(xb, wihT, bias, u, bl_ih, gxbuf, kI);
  k_lstm_scan<<<2, 512, smem, stream>>>(gxbuf, whhT, a_hh, bl_hh, ybf,
                                        (float*)nullptr, hn, cn);

  // ---- layer 1 (cells 2,3): input = layer-0 output (ybf) ----
  k_lora_u<<<(2 * kBT * kR) / 256, 256, 0, stream>>>(
      (const float*)nullptr, ybf, 1, a_ih + (size_t)2 * kR * kI,
      u + (size_t)2 * kBT * kR, kI);
  k_gx_gemm<<<8192, 256, 0, stream>>>(
      ybf, wihT + (size_t)2 * kI * kG, bias + (size_t)2 * kG,
      u + (size_t)2 * kBT * kR, bl_ih + (size_t)2 * kG * kR, gxbuf, kI);
  k_lstm_scan<<<2, 512, smem, stream>>>(
      gxbuf, whhT + (size_t)2 * kH * kG, a_hh + (size_t)2 * kR * kH,
      bl_hh + (size_t)2 * kG * kR, ybf, out32,
      hn + (size_t)2 * kB * kH, cn + (size_t)2 * kB * kH);
}
